// CrossAttention_86174223827377
// MI455X (gfx1250) — matrix operand order repaired and checked by reading, not compiled, not run on hardware
//
#include <hip/hip_runtime.h>

using half_t = _Float16;
typedef __attribute__((ext_vector_type(16))) _Float16 v16h;
typedef __attribute__((ext_vector_type(8)))  _Float16 v8h;
typedef __attribute__((ext_vector_type(4)))  _Float16 v4h;
typedef __attribute__((ext_vector_type(8)))  float    v8f;
typedef __attribute__((ext_vector_type(4)))  float    v4f;
typedef __attribute__((ext_vector_type(4)))  unsigned int u32x4;
typedef __attribute__((ext_vector_type(8)))  int      i32x8;
typedef __attribute__((ext_vector_type(4)))  int      i32x4;

union AFrag { v16h v; v8h h[2]; };

__device__ __forceinline__ v8f wmma_f16(v16h a, v16h b, v8f c) {
  return __builtin_amdgcn_wmma_f32_16x16x32_f16(
      false, a, false, b, (short)0, c, false, false);
}

// ---------------------------------------------------------------------------
// TDM 2D tile load global->LDS with hardware LDS padding (D# per ISA ch.8).
// pad_int_enc: pad after (2<<enc) DWORDs ; pad_amt_enc: pad of (enc+1) DWORDs.
// ---------------------------------------------------------------------------
#if __has_builtin(__builtin_amdgcn_tensor_load_to_lds)
#define HAVE_TDM 1
__device__ __forceinline__ void tdm_load_2d(unsigned lds_byte_off, const void* gaddr,
                                            unsigned elems_per_row, unsigned rows,
                                            unsigned row_stride_elems,
                                            unsigned pad_int_enc, unsigned pad_amt_enc)
{
  unsigned long long ga = (unsigned long long)(uintptr_t)gaddr;
  u32x4 g0;
  g0[0] = 1u;                                            // count=1 (valid descriptor)
  g0[1] = lds_byte_off;                                  // lds_addr
  g0[2] = (unsigned)(ga & 0xffffffffu);                  // global_addr[31:0]
  g0[3] = (unsigned)((ga >> 32) & 0x01ffffffu) | (2u << 30);  // addr[56:32] | type=2
  i32x8 g1;
  g1[0] = (int)((1u << 16)                               // data_size = 2 bytes
              | (1u << 20)                               // pad_enable
              | (pad_int_enc << 22) | (pad_amt_enc << 25));
  g1[1] = (int)((elems_per_row & 0xffffu) << 16);        // tensor_dim0[15:0]
  g1[2] = (int)(((elems_per_row >> 16) & 0xffffu) | ((rows & 0xffffu) << 16));
  g1[3] = (int)(((rows >> 16) & 0xffffu) | ((elems_per_row & 0xffffu) << 16)); // tile_dim0
  g1[4] = (int)(rows & 0xffffu);                         // tile_dim1 (tile_dim2=0)
  g1[5] = (int)row_stride_elems;                         // tensor_dim0_stride[31:0]
  g1[6] = 0;
  g1[7] = 0;
  i32x4 g2 = {0, 0, 0, 0};
  i32x4 g3 = {0, 0, 0, 0};
#if defined(__clang_major__) && (__clang_major__ >= 23)
  i32x8 g4 = {0, 0, 0, 0, 0, 0, 0, 0};
  __builtin_amdgcn_tensor_load_to_lds(g0, g1, g2, g3, g4, 0);
#else
  __builtin_amdgcn_tensor_load_to_lds(g0, g1, g2, g3, 0);
#endif
}
#else
#define HAVE_TDM 0
#endif

// ---------------------------------------------------------------------------
// GEMM: C[M,512] = A[M,K] @ B[K,512], double-buffered LDS, 128x128x32 tiles,
// 8 waves (4x2), each wave 32x64 (2x4 WMMA tiles of 16x16x32 f16).
// MODE 0: A f32 -> f16 out, head-split [b*8+h][seq][64], scaled by out_scale
// MODE 1: A f16 -> f32 out row-major + bias
// MODE 2: A f32 -> f16 out, V-transposed [b*8+h][64 d][512 pos], * out_scale
// ---------------------------------------------------------------------------
template <int MODE>
__global__ __launch_bounds__(256)
void gemm512(const void* __restrict__ Ap, const float* __restrict__ B,
             void* __restrict__ Cp, const float* __restrict__ bias,
             int M, int K, int seq_shift, float out_scale)
{
  __shared__ __align__(16) half_t As[2][128 * 40];   // [row][k], stride 40 halfs
  __shared__ __align__(16) half_t Bs[2][128 * 40];   // transposed [n][k]

  const int tid  = threadIdx.x;
  const int bm   = blockIdx.x * 128;
  const int bn   = blockIdx.y * 128;
  const int w    = tid >> 5;
  const int lane = tid & 31;
  const int l16  = lane & 15;
  const int hs   = lane >> 4;
  const int wm   = (w & 3) * 32;
  const int wn   = (w >> 2) * 64;

  v8f acc[2][4];
  for (int mt = 0; mt < 2; mt++)
    for (int nt = 0; nt < 4; nt++)
      for (int i = 0; i < 8; i++) acc[mt][nt][i] = 0.f;

  auto fill = [&](half_t* As_, half_t* Bs_, int k0) {
    if constexpr (MODE != 1) {
      const float* A = (const float*)Ap;
      for (int j = 0; j < 4; j++) {
        int id = tid + j * 256;
        int row = id >> 3, c4 = id & 7;
        v4f f = *(const v4f*)(A + (size_t)(bm + row) * K + k0 + c4 * 4);
        v4h hh;
        hh[0] = (half_t)f[0]; hh[1] = (half_t)f[1];
        hh[2] = (half_t)f[2]; hh[3] = (half_t)f[3];
        *(v4h*)&As_[row * 40 + c4 * 4] = hh;
        if (k0 + 64 < K)
          __builtin_prefetch(A + (size_t)(bm + row) * K + k0 + 64 + c4 * 4, 0, 1);
      }
    } else {
      const half_t* A = (const half_t*)Ap;
      for (int j = 0; j < 2; j++) {
        int id = tid + j * 256;
        int row = id >> 2, c8 = id & 3;
        *(v8h*)&As_[row * 40 + c8 * 8] =
            *(const v8h*)(A + (size_t)(bm + row) * K + k0 + c8 * 8);
        if (k0 + 64 < K)
          __builtin_prefetch(A + (size_t)(bm + row) * K + k0 + 64 + c8 * 8, 0, 1);
      }
    }
    for (int j = 0; j < 4; j++) {
      int id = tid + j * 256;
      int kr = id >> 5, c4 = id & 31;
      v4f f = *(const v4f*)(B + (size_t)(k0 + kr) * 512 + bn + c4 * 4);
      for (int u = 0; u < 4; u++)
        Bs_[(c4 * 4 + u) * 40 + kr] = (half_t)f[u];
    }
  };

  auto compute = [&](const half_t* As_, const half_t* Bs_) {
    AFrag af[2];
    for (int mt = 0; mt < 2; mt++) {
      int row = wm + mt * 16 + l16;
      int kb  = hs * 8;
      af[mt].h[0] = *(const v8h*)&As_[row * 40 + kb];
      af[mt].h[1] = *(const v8h*)&As_[row * 40 + kb + 16];
    }
    for (int nt = 0; nt < 4; nt++) {
      AFrag bf;
      int col = wn + nt * 16 + l16;
      int ko  = hs * 8;
      bf.h[0] = *(const v8h*)&Bs_[col * 40 + ko];
      bf.h[1] = *(const v8h*)&Bs_[col * 40 + ko + 16];
      for (int mt = 0; mt < 2; mt++)
        acc[mt][nt] = wmma_f16(af[mt].v, bf.v, acc[mt][nt]);
    }
  };

  const int nk = K >> 5;
  fill(As[0], Bs[0], 0);
  int cur = 0;
  for (int it = 0; it < nk; ++it) {
    __syncthreads();
    if (it + 1 < nk) fill(As[cur ^ 1], Bs[cur ^ 1], (it + 1) * 32);
    compute(As[cur], Bs[cur]);
    cur ^= 1;
  }

  for (int mt = 0; mt < 2; mt++)
    for (int nt = 0; nt < 4; nt++)
      for (int i = 0; i < 8; i++) {
        int gm = bm + wm + mt * 16 + i + 8 * hs;
        int gn = bn + wn + nt * 16 + l16;
        float val = acc[mt][nt][i];
        if constexpr (MODE == 0) {
          int b_idx = gm >> seq_shift;
          int pos   = gm & ((1 << seq_shift) - 1);
          int h_idx = gn >> 6, d = gn & 63;
          ((half_t*)Cp)[(((size_t)(b_idx * 8 + h_idx) << seq_shift) + pos) * 64 + d] =
              (half_t)(val * out_scale);
        } else if constexpr (MODE == 2) {
          int b_idx = gm >> 9;                 // seq = 512 (context)
          int pos   = gm & 511;
          int h_idx = gn >> 6, d = gn & 63;
          ((half_t*)Cp)[((size_t)((b_idx * 8 + h_idx) * 64 + d) << 9) + pos] =
              (half_t)(val * out_scale);
        } else {
          ((float*)Cp)[(size_t)gm * 512 + gn] = val + bias[gn];
        }
      }
}

// ---------------------------------------------------------------------------
// Flash attention: block = (128-row q tile, bh). Q pre-scaled by 1/8 in proj.
// Q[128][64], K[512][64], Vt[64][512] DMA'd into padded LDS by the TDM.
// ---------------------------------------------------------------------------
__global__ __launch_bounds__(256)
void attn_kernel(const half_t* __restrict__ q, const half_t* __restrict__ k,
                 const half_t* __restrict__ v, half_t* __restrict__ o)
{
  extern __shared__ __align__(16) half_t smem[];
  half_t* Qs = smem;                 // [128][72]
  half_t* Ks = Qs + 128 * 72;        // [512][72]
  half_t* Vt = Ks + 512 * 72;        // [64][520]
  half_t* Pb = Vt + 64 * 520;        // 8 waves x [16][72]

  const int tid  = threadIdx.x;
  const int qt   = blockIdx.x;
  const int bh   = blockIdx.y;
  const int w    = tid >> 5;
  const int lane = tid & 31;
  const int l16  = lane & 15;
  const int hs   = lane >> 4;

  const half_t* qp = q + ((size_t)bh * 4096 + (size_t)qt * 128) * 64;
  const half_t* kp = k + (size_t)bh * 512 * 64;
  const half_t* vp = v + (size_t)bh * 64 * 512;   // pre-transposed [64 d][512 pos]

#if HAVE_TDM
  {
    const unsigned base = __builtin_amdgcn_groupstaticsize();
    // pad encodings: interval (2<<enc) DWORDs, amount (enc+1) DWORDs
    if (w == 0) {        // Q: 128 rows x 64 halfs, +8 halfs pad -> stride 72
      tdm_load_2d(base + 0,      qp, 64, 128, 64, /*32 dw*/4, /*4 dw*/3);
      __builtin_amdgcn_s_wait_tensorcnt(0);
    } else if (w == 1) { // K: 512 rows x 64 halfs -> stride 72
      tdm_load_2d(base + 18432,  kp, 64, 512, 64, 4, 3);
      __builtin_amdgcn_s_wait_tensorcnt(0);
    } else if (w == 2) { // Vt: 64 rows x 512 halfs, +8 halfs pad -> stride 520
      tdm_load_2d(base + 92160,  vp, 512, 64, 512, /*256 dw*/7, 3);
      __builtin_amdgcn_s_wait_tensorcnt(0);
    }
  }
#else
  for (int j = 0; j < 4; j++) {
    int id = tid + j * 256;
    int row = id >> 3, c = (id & 7) * 8;
    *(v8h*)&Qs[row * 72 + c] = *(const v8h*)(qp + (size_t)row * 64 + c);
  }
  for (int j = 0; j < 16; j++) {
    int id = tid + j * 256;
    int row = id >> 3, c = (id & 7) * 8;
    *(v8h*)&Ks[row * 72 + c] = *(const v8h*)(kp + (size_t)row * 64 + c);
  }
  for (int j = 0; j < 16; j++) {
    int id = tid + j * 256;
    int d = id >> 6, c = (id & 63) * 8;
    *(v8h*)&Vt[d * 520 + c] = *(const v8h*)(vp + (size_t)d * 512 + c);
  }
#endif
  __syncthreads();

  half_t* Pw = Pb + w * 16 * 72;
  const int qr0 = w * 16;

  float mstate[8], lstate[8];
  v8f oacc[4];
  for (int i = 0; i < 8; i++) { mstate[i] = -1e30f; lstate[i] = 0.f; }
  for (int nt = 0; nt < 4; nt++)
    for (int i = 0; i < 8; i++) oacc[nt][i] = 0.f;

  for (int c = 0; c < 8; c++) {
    const int cm = c * 64;
    v8f s[4];
    for (int nt = 0; nt < 4; nt++)
      for (int i = 0; i < 8; i++) s[nt][i] = 0.f;
    for (int kk = 0; kk < 64; kk += 32) {
      AFrag qa;
      int ar = qr0 + l16, kb = kk + hs * 8;
      qa.h[0] = *(const v8h*)&Qs[ar * 72 + kb];
      qa.h[1] = *(const v8h*)&Qs[ar * 72 + kb + 16];
      for (int nt = 0; nt < 4; nt++) {
        AFrag kf;
        int col = cm + nt * 16 + l16;
        int ko  = kk + hs * 8;
        kf.h[0] = *(const v8h*)&Ks[col * 72 + ko];
        kf.h[1] = *(const v8h*)&Ks[col * 72 + ko + 16];
        s[nt] = wmma_f16(qa.v, kf.v, s[nt]);
      }
    }
    float rm[8];
    for (int i = 0; i < 8; i++)
      rm[i] = fmaxf(fmaxf(s[0][i], s[1][i]), fmaxf(s[2][i], s[3][i]));
    for (int off = 1; off < 16; off <<= 1)
      for (int i = 0; i < 8; i++)
        rm[i] = fmaxf(rm[i], __shfl_xor(rm[i], off, 32));
    float sc[8];
    for (int i = 0; i < 8; i++) {
      float mn = fmaxf(mstate[i], rm[i]);
      sc[i] = __expf(mstate[i] - mn);
      mstate[i] = mn;
    }
    for (int nt = 0; nt < 4; nt++)
      for (int i = 0; i < 8; i++)
        s[nt][i] = __expf(s[nt][i] - mstate[i]);
    float rs[8];
    for (int i = 0; i < 8; i++) rs[i] = s[0][i] + s[1][i] + s[2][i] + s[3][i];
    for (int off = 1; off < 16; off <<= 1)
      for (int i = 0; i < 8; i++) rs[i] += __shfl_xor(rs[i], off, 32);
    for (int i = 0; i < 8; i++) lstate[i] = lstate[i] * sc[i] + rs[i];

    for (int nt = 0; nt < 4; nt++)
      for (int i = 0; i < 8; i++)
        Pw[(i + 8 * hs) * 72 + nt * 16 + l16] = (half_t)s[nt][i];
    asm volatile("s_wait_dscnt 0x0" ::: "memory");

    for (int nt = 0; nt < 4; nt++)
      for (int i = 0; i < 8; i++) oacc[nt][i] *= sc[i];

    for (int kk = 0; kk < 64; kk += 32) {
      AFrag pa;
      int kb = kk + hs * 8;
      pa.h[0] = *(const v8h*)&Pw[l16 * 72 + kb];
      pa.h[1] = *(const v8h*)&Pw[l16 * 72 + kb + 16];
      for (int nt = 0; nt < 4; nt++) {
        AFrag vf;
        int dcol = nt * 16 + l16;
        int ko   = cm + kk + hs * 8;
        vf.h[0] = *(const v8h*)&Vt[dcol * 520 + ko];
        vf.h[1] = *(const v8h*)&Vt[dcol * 520 + ko + 16];
        oacc[nt] = wmma_f16(pa.v, vf.v, oacc[nt]);
      }
    }
  }

  float inv[8];
  for (int i = 0; i < 8; i++) inv[i] = 1.0f / lstate[i];
  const int b_idx = bh >> 3, h_idx = bh & 7;
  for (int nt = 0; nt < 4; nt++)
    for (int i = 0; i < 8; i++) {
      int grow = b_idx * 4096 + qt * 128 + qr0 + i + 8 * hs;
      int gcol = h_idx * 64 + nt * 16 + l16;
      o[(size_t)grow * 512 + gcol] = (half_t)(oacc[nt][i] * inv[i]);
    }
}

// ---------------------------------------------------------------------------
extern "C" void kernel_launch(void* const* d_in, const int* in_sizes, int n_in,
                              void* d_out, int out_size, void* d_ws, size_t ws_size,
                              hipStream_t stream)
{
  (void)in_sizes; (void)n_in; (void)out_size; (void)ws_size;
  const float* x     = (const float*)d_in[0];   // [8,4096,512]
  const float* ctx   = (const float*)d_in[1];   // [8,512,768]
  const float* w_q   = (const float*)d_in[2];   // [512,512]
  const float* w_k   = (const float*)d_in[3];   // [768,512]
  const float* w_v   = (const float*)d_in[4];   // [768,512]
  const float* w_out = (const float*)d_in[5];   // [512,512]
  const float* b_out = (const float*)d_in[6];   // [512]

  half_t* qb = (half_t*)d_ws;                      // [64][4096][64] f16 (pre-scaled)
  half_t* kb = qb + (size_t)64 * 4096 * 64;        // [64][512][64]
  half_t* vb = kb + (size_t)64 * 512 * 64;         // [64][64][512] (transposed)
  half_t* ob = vb + (size_t)64 * 512 * 64;         // [32768][512] f16

  gemm512<0><<<dim3(256, 4), 256, 0, stream>>>(x,   w_q, qb, nullptr, 32768, 512, 12, 0.125f);
  gemm512<0><<<dim3(32, 4),  256, 0, stream>>>(ctx, w_k, kb, nullptr, 4096,  768, 9,  1.0f);
  gemm512<2><<<dim3(32, 4),  256, 0, stream>>>(ctx, w_v, vb, nullptr, 4096,  768, 9,  1.0f);

  constexpr size_t attn_lds =
      (size_t)(128 * 72 + 512 * 72 + 64 * 520 + 8 * 16 * 72) * sizeof(half_t);
  attn_kernel<<<dim3(32, 64), 256, attn_lds, stream>>>(qb, kb, vb, ob);

  gemm512<1><<<dim3(256, 4), 256, 0, stream>>>(ob, w_out, (float*)d_out, b_out, 32768, 512, 0, 1.0f);
}
